// GraphAttentionLayer_51384988729608
// MI455X (gfx1250) — matrix operand order repaired and checked by reading, not compiled, not run on hardware
//
#include <hip/hip_runtime.h>
#include <hip/hip_bf16.h>
#include <math.h>

#define N_NODES 50000
#define N_EDGES 1600000
#define IN_FEAT 256
#define OUT_FEAT 128
#define LRELU_ALPHA 0.2f

typedef __attribute__((ext_vector_type(16))) __bf16 v16bf;
typedef __attribute__((ext_vector_type(8)))  float  v8f;

// ---------- monotonic float <-> uint encoding for atomicMax over floats ----------
__device__ __forceinline__ unsigned enc_f32(float f) {
    unsigned u = __float_as_uint(f);
    return (u & 0x80000000u) ? ~u : (u | 0x80000000u);
}
__device__ __forceinline__ float dec_f32(unsigned u) {
    return (u & 0x80000000u) ? __uint_as_float(u & 0x7FFFFFFFu) : __uint_as_float(~u);
}

// ---------- 0: init scratch (accum, denom, max) ----------
__global__ void gat_init(float* __restrict__ accum, float* __restrict__ denom,
                         unsigned* __restrict__ maxenc) {
    int i = blockIdx.x * blockDim.x + threadIdx.x;
    if (i < N_NODES * OUT_FEAT) accum[i] = 0.0f;
    if (i < N_NODES) { denom[i] = 0.0f; maxenc[i] = 0u; }  // 0u sits below every encoded real
}

// ---------- 0b: pack W into WMMA B-fragment order (bf16, 64 KB) ----------
// Wp[((tn*8 + kb)*32 + lane)*16 + j] = bf16( W[kb*32 + (lane>>4)*8 + (j&7) + (j>>3)*16][tn*16 + (lane&15)] )
__global__ void gat_pack_w(const float* __restrict__ W, __bf16* __restrict__ Wp) {
    int t = blockIdx.x * blockDim.x + threadIdx.x;             // 0 .. 32767
    if (t >= (IN_FEAT * OUT_FEAT)) return;
    int j    = t & 15;
    int lane = (t >> 4) & 31;
    int kb   = (t >> 9) & 7;
    int tn   = t >> 12;
    int k = kb * 32 + (lane >> 4) * 8 + (j & 7) + (j >> 3) * 16;
    int n = tn * 16 + (lane & 15);
    Wp[t] = (__bf16)W[(size_t)k * OUT_FEAT + n];
}

// ---------- A-fragment loader: 16x32 bf16 layout from f32 rows ----------
// lane half0 -> K {0..7} U {16..23}; half1 -> K {8..15} U {24..31}
__device__ __forceinline__ v16bf load_a_frag(const float* __restrict__ base) {
    v16bf a;
    #pragma unroll
    for (int i = 0; i < 8; ++i) {
        a[i]     = (__bf16)base[i];
        a[8 + i] = (__bf16)base[16 + i];
    }
    return a;
}

// ---------- 1: Wh = h @ W via v_wmma_f32_16x16x32_bf16 ----------
// 8 waves per block, one 16x16 output tile per wave; dual accumulators for
// two independent WMMA chains (co-execute with loads/converts).
__global__ __launch_bounds__(256) void gat_gemm_wmma(const float* __restrict__ h,
                                                     const __bf16* __restrict__ Wp,
                                                     float* __restrict__ Wh) {
    const int tiles_n = OUT_FEAT / 16;                       // 8
    const int wave    = (blockIdx.x * blockDim.x + threadIdx.x) >> 5;
    const int lane    = threadIdx.x & 31;
    const int tm      = wave / tiles_n;                      // 0..3124
    const int tn      = wave % tiles_n;                      // 0..7
    if (tm >= N_NODES / 16) return;

    const int half = lane >> 4;
    const int m    = lane & 15;
    const int row  = tm * 16 + m;
    const int col  = tn * 16 + m;

    const float*  arow = h + (size_t)row * IN_FEAT + half * 8;
    const __bf16* bptr = Wp + (((size_t)tn * 8) * 32 + lane) * 16;

    v8f c0 = {}, c1 = {};
    #pragma unroll
    for (int kb = 0; kb < IN_FEAT / 32; kb += 2) {
        v16bf a0 = load_a_frag(arow + kb * 32);
        v16bf b0 = *(const v16bf*)(bptr + (size_t)kb * 32 * 16);
        c0 = __builtin_amdgcn_wmma_f32_16x16x32_bf16(false, a0, false, b0,
                                                     (short)0, c0, false, false);
        v16bf a1 = load_a_frag(arow + (kb + 1) * 32);
        v16bf b1 = *(const v16bf*)(bptr + (size_t)(kb + 1) * 32 * 16);
        c1 = __builtin_amdgcn_wmma_f32_16x16x32_bf16(false, a1, false, b1,
                                                     (short)0, c1, false, false);
    }
    c0 = c0 + c1;

    // C/D: VGPR j -> row (half ? 8+j : j), N = lane%16
    #pragma unroll
    for (int j = 0; j < 8; ++j) {
        int r = tm * 16 + (half ? (8 + j) : j);
        Wh[(size_t)r * OUT_FEAT + col] = c0[j];
    }
}

// ---------- 2: f1 = Wh @ a_src, f2 = Wh @ a_dst (one thread per node) ----------
__global__ void gat_scores(const float* __restrict__ Wh,
                           const float* __restrict__ a_src,
                           const float* __restrict__ a_dst,
                           float* __restrict__ f1, float* __restrict__ f2) {
    int i = blockIdx.x * blockDim.x + threadIdx.x;
    if (i >= N_NODES) return;
    const float4* row = (const float4*)(Wh + (size_t)i * OUT_FEAT);
    const float4* as  = (const float4*)a_src;
    const float4* ad  = (const float4*)a_dst;
    float s1 = 0.0f, s2 = 0.0f;
    #pragma unroll
    for (int j = 0; j < OUT_FEAT / 4; ++j) {
        float4 v = row[j], x = as[j], y = ad[j];
        s1 += v.x * x.x + v.y * x.y + v.z * x.z + v.w * x.w;
        s2 += v.x * y.x + v.y * y.y + v.z * y.z + v.w * y.w;
    }
    f1[i] = s1;
    f2[i] = s2;
}

// ---------- 3: per-edge score + segment max ----------
__global__ void gat_edge_max(const int* __restrict__ src, const int* __restrict__ dst,
                             const float* __restrict__ f1, const float* __restrict__ f2,
                             float* __restrict__ ebuf, unsigned* __restrict__ maxenc) {
    int i = blockIdx.x * blockDim.x + threadIdx.x;
    if (i >= N_EDGES) return;
    int s = src[i], d = dst[i];
    float e = f1[s] + f2[d];
    e = (e > 0.0f) ? e : (LRELU_ALPHA * e);                  // LeakyReLU
    ebuf[i] = e;
    atomicMax(&maxenc[s], enc_f32(e));
}

// ---------- 4: exp(e - max) + segment sum ----------
__global__ void gat_edge_exp(const int* __restrict__ src,
                             float* __restrict__ ebuf,
                             const unsigned* __restrict__ maxenc,
                             float* __restrict__ denom) {
    int i = blockIdx.x * blockDim.x + threadIdx.x;
    if (i >= N_EDGES) return;
    int s = src[i];
    float ex = __expf(ebuf[i] - dec_f32(maxenc[s]));
    ebuf[i] = ex;
    unsafeAtomicAdd(&denom[s], ex);                          // global_atomic_add_f32
}

// ---------- 5: h_prime[src] += att * Wh[dst]  (one wave per edge) ----------
// Wave-uniform metadata forced into SGPRs via readfirstlane -> scalar loads;
// VMEM pipe reserved for the 512B row gather + f32 atomics.
__global__ __launch_bounds__(256) void gat_aggregate(const int* __restrict__ src,
                                                     const int* __restrict__ dst,
                                                     const float* __restrict__ ebuf,
                                                     const float* __restrict__ denom,
                                                     const float* __restrict__ Wh,
                                                     float* __restrict__ accum) {
    int edge = (blockIdx.x * blockDim.x + threadIdx.x) >> 5;
    int lane = threadIdx.x & 31;
    if (edge >= N_EDGES) return;

    int eu = __builtin_amdgcn_readfirstlane(edge);           // wave-uniform -> SGPR
    int s  = __builtin_amdgcn_readfirstlane(src[eu]);
    int d  = __builtin_amdgcn_readfirstlane(dst[eu]);
    float att = ebuf[eu] / denom[s];

    float4 v  = ((const float4*)(Wh + (size_t)d * OUT_FEAT))[lane];
    float* out = accum + (size_t)s * OUT_FEAT + lane * 4;
    unsafeAtomicAdd(out + 0, att * v.x);
    unsafeAtomicAdd(out + 1, att * v.y);
    unsafeAtomicAdd(out + 2, att * v.z);
    unsafeAtomicAdd(out + 3, att * v.w);
}

// ---------- 6: ELU into d_out ----------
__global__ void gat_elu(const float* __restrict__ accum, float* __restrict__ out) {
    int i = blockIdx.x * blockDim.x + threadIdx.x;
    if (i >= N_NODES * OUT_FEAT) return;
    float x = accum[i];
    out[i] = (x > 0.0f) ? x : (__expf(x) - 1.0f);
}

extern "C" void kernel_launch(void* const* d_in, const int* in_sizes, int n_in,
                              void* d_out, int out_size, void* d_ws, size_t ws_size,
                              hipStream_t stream) {
    // setup_inputs order: h, src, dst, W, a_src, a_dst
    const float* h     = (const float*)d_in[0];
    const int*   src   = (const int*)d_in[1];
    const int*   dst   = (const int*)d_in[2];
    const float* W     = (const float*)d_in[3];
    const float* a_src = (const float*)d_in[4];
    const float* a_dst = (const float*)d_in[5];
    float*       out   = (float*)d_out;

    // workspace carve-up (Wp first for 32B vector-load alignment)
    char* ws = (char*)d_ws;
    size_t off = 0;
    __bf16*   Wp     = (__bf16*)(ws + off);   off += (size_t)IN_FEAT * OUT_FEAT * sizeof(__bf16);
    float*    Wh     = (float*)(ws + off);    off += (size_t)N_NODES * OUT_FEAT * sizeof(float);
    float*    f1     = (float*)(ws + off);    off += (size_t)N_NODES * sizeof(float);
    float*    f2     = (float*)(ws + off);    off += (size_t)N_NODES * sizeof(float);
    unsigned* maxenc = (unsigned*)(ws + off); off += (size_t)N_NODES * sizeof(unsigned);
    float*    denom  = (float*)(ws + off);    off += (size_t)N_NODES * sizeof(float);
    float*    ebuf   = (float*)(ws + off);    off += (size_t)N_EDGES * sizeof(float);
    float*    accum  = (float*)(ws + off);    off += (size_t)N_NODES * OUT_FEAT * sizeof(float);
    (void)ws_size; (void)in_sizes; (void)n_in; (void)out_size;

    const int elems = N_NODES * OUT_FEAT;

    // 0: zero accumulators / softmax stats; pack W to bf16 fragments
    gat_init<<<(elems + 255) / 256, 256, 0, stream>>>(accum, denom, maxenc);
    gat_pack_w<<<(IN_FEAT * OUT_FEAT + 255) / 256, 256, 0, stream>>>(W, Wp);

    // 1: Wh = h @ W (WMMA). 3125*8 = 25000 tiles, 8 waves/block -> 3125 blocks
    gat_gemm_wmma<<<(N_NODES / 16) * (OUT_FEAT / 16) / 8, 256, 0, stream>>>(h, Wp, Wh);

    // 2: attention score projections
    gat_scores<<<(N_NODES + 255) / 256, 256, 0, stream>>>(Wh, a_src, a_dst, f1, f2);

    // 3: edge scores + segment max
    gat_edge_max<<<(N_EDGES + 255) / 256, 256, 0, stream>>>(src, dst, f1, f2, ebuf, maxenc);

    // 4: exp + segment sum
    gat_edge_exp<<<(N_EDGES + 255) / 256, 256, 0, stream>>>(src, ebuf, maxenc, denom);

    // 5: weighted aggregation, one wave per edge
    gat_aggregate<<<(N_EDGES * 32 + 255) / 256, 256, 0, stream>>>(src, dst, ebuf, denom, Wh, accum);

    // 6: ELU -> d_out
    gat_elu<<<(elems + 255) / 256, 256, 0, stream>>>(accum, out);
}
